// RfLegoBeamformerModel_15857019257619
// MI455X (gfx1250) — matrix operand order repaired and checked by reading, not compiled, not run on hardware
//
#include <hip/hip_runtime.h>
#include <hip/hip_bf16.h>
#include <math.h>

#define L_LAYERS 10
#define NDIM 256
#define MDIM 64
#define BATCH 16384
#define BN_EPS 1e-5f

typedef __attribute__((ext_vector_type(16))) __bf16 v16bf;
typedef __attribute__((ext_vector_type(8)))  __bf16 v8bf;
typedef __attribute__((ext_vector_type(4)))  __bf16 v4bf;
typedef __attribute__((ext_vector_type(8)))  float  v8f;
typedef __attribute__((ext_vector_type(4)))  float  v4f;

__device__ __forceinline__ float sigm(float x) { return 1.0f / (1.0f + __expf(-x)); }
__device__ __forceinline__ float softplus(float x) { return x > 20.0f ? x : log1pf(__expf(x)); }

__device__ __forceinline__ v8f wmma_bf16(v16bf a, v16bf b, v8f c) {
  return __builtin_amdgcn_wmma_f32_16x16x32_bf16(false, a, false, b, (short)0, c, false, false);
}

// ---------------------------------------------------------------------------
// Kernel 1: A^H y. 2.1 GB of A streamed exactly once -> nontemporal loads so
// the L2 stays reserved for weights/activations. v4f per lane, 4 samples per
// 256-thread block.
// ---------------------------------------------------------------------------
__global__ void ahy_kernel(const float* __restrict__ yr, const float* __restrict__ yi,
                           const float* __restrict__ Ar, const float* __restrict__ Ai,
                           float* __restrict__ outr, float* __restrict__ outi) {
  __shared__ float s_yr[4][MDIM], s_yi[4][MDIM];
  const int tid = threadIdx.x;
  const int sub = tid >> 6;    // sample within block (0..3)
  const int t64 = tid & 63;
  const int b = blockIdx.x * 4 + sub;
  s_yr[sub][t64] = yr[(size_t)b * MDIM + t64];
  s_yi[sub][t64] = yi[(size_t)b * MDIM + t64];
  __syncthreads();
  const int n4 = t64 * 4;
  const v4f* ar = (const v4f*)(Ar + (size_t)b * MDIM * NDIM + n4);
  const v4f* ai = (const v4f*)(Ai + (size_t)b * MDIM * NDIM + n4);
  v4f accr = {0.f, 0.f, 0.f, 0.f}, acci = {0.f, 0.f, 0.f, 0.f};
#pragma unroll 4
  for (int m = 0; m < MDIM; ++m) {
    v4f a = __builtin_nontemporal_load(&ar[(size_t)m * (NDIM / 4)]);
    v4f c = __builtin_nontemporal_load(&ai[(size_t)m * (NDIM / 4)]);
    const float u = s_yr[sub][m], v = s_yi[sub][m];
#pragma unroll
    for (int e = 0; e < 4; ++e) {
      accr[e] = fmaf(a[e], u, fmaf(c[e], v, accr[e]));
      acci[e] = fmaf(a[e], v, fmaf(-c[e], u, acci[e]));
    }
  }
  *(v4f*)(outr + (size_t)b * NDIM + n4) = accr;
  *(v4f*)(outi + (size_t)b * NDIM + n4) = acci;
}

// ---------------------------------------------------------------------------
// fp32 -> bf16 mirror for one [L*N*N] weight tensor (one-time, 2.6 MB/matrix)
// ---------------------------------------------------------------------------
__global__ void f32_to_bf16_kernel(const float* __restrict__ src, __bf16* __restrict__ dst) {
  const size_t i = ((size_t)blockIdx.x * blockDim.x + threadIdx.x) * 4;
  v4f x = *(const v4f*)(src + i);
  v4bf o;
  o[0] = (__bf16)x[0]; o[1] = (__bf16)x[1]; o[2] = (__bf16)x[2]; o[3] = (__bf16)x[3];
  *(v4bf*)(dst + i) = o;
}

// ---------------------------------------------------------------------------
// BatchNorm stats / finalize / apply (+ zero z,v masters and bf16 mirrors)
// ---------------------------------------------------------------------------
__global__ void stats_init_kernel(float* __restrict__ stats) {
  const int t = threadIdx.x;
#pragma unroll
  for (int j = 0; j < 8; ++j) stats[j * NDIM + t] = 0.0f;
}

__global__ void bn_partial_kernel(const float* __restrict__ r, const float* __restrict__ i,
                                  float* __restrict__ stats) {
  const int n = threadIdx.x;
  const int rows = BATCH / 64;  // gridDim.x == 64
  float sr = 0.f, qr = 0.f, si = 0.f, qi = 0.f;
  for (int k = 0; k < rows; ++k) {
    size_t idx = (size_t)(blockIdx.x * rows + k) * NDIM + n;
    float a = r[idx]; sr += a; qr = fmaf(a, a, qr);
    float c = i[idx]; si += c; qi = fmaf(c, c, qi);
  }
  atomicAdd(&stats[0 * NDIM + n], sr);
  atomicAdd(&stats[1 * NDIM + n], qr);
  atomicAdd(&stats[2 * NDIM + n], si);
  atomicAdd(&stats[3 * NDIM + n], qi);
}

__global__ void bn_finalize_kernel(const float* __restrict__ stats, float* __restrict__ ms) {
  const int n = threadIdx.x;
  const float invB = 1.0f / (float)BATCH;
  float mr = stats[0 * NDIM + n] * invB;
  float vr = stats[1 * NDIM + n] * invB - mr * mr;
  float mi = stats[2 * NDIM + n] * invB;
  float vi = stats[3 * NDIM + n] * invB - mi * mi;
  ms[0 * NDIM + n] = mr;
  ms[1 * NDIM + n] = rsqrtf(vr + BN_EPS);
  ms[2 * NDIM + n] = mi;
  ms[3 * NDIM + n] = rsqrtf(vi + BN_EPS);
}

__global__ void bn_apply_kernel(float* __restrict__ r, float* __restrict__ i,
                                const float* __restrict__ ms,
                                float* __restrict__ zr, float* __restrict__ zi,
                                float* __restrict__ vr, float* __restrict__ vi,
                                __bf16* __restrict__ zbr, __bf16* __restrict__ zbi,
                                __bf16* __restrict__ vbr, __bf16* __restrict__ vbi) {
  const size_t idx = (size_t)blockIdx.x * blockDim.x + threadIdx.x;
  const int n = (int)(idx & (NDIM - 1));
  r[idx] = (r[idx] - ms[n]) * ms[NDIM + n];
  i[idx] = (i[idx] - ms[2 * NDIM + n]) * ms[3 * NDIM + n];
  zr[idx] = 0.f; zi[idx] = 0.f; vr[idx] = 0.f; vi[idx] = 0.f;
  const __bf16 z = (__bf16)0.f;
  zbr[idx] = z; zbi[idx] = z; vbr[idx] = z; vbi[idx] = z;
}

// ---------------------------------------------------------------------------
// One scan layer: bf16 WMMA gate GEMMs + fused complex ADMM update.
// Block = 8 waves, owns 16 batch rows x all 256 cols. Activation panels
// (z_r, v_r, z_i, v_i as bf16) staged once into 32 KB LDS; A-frags come from
// ds_load_b128, B-frags from L2-resident bf16 weights. Epilogue updates fp32
// masters in place and refreshes the bf16 mirrors for the next layer.
// ---------------------------------------------------------------------------
__global__ __launch_bounds__(256) void layer_kernel(
    float* __restrict__ zr, float* __restrict__ zi,
    float* __restrict__ vr, float* __restrict__ vi,
    __bf16* __restrict__ zbr, __bf16* __restrict__ zbi,
    __bf16* __restrict__ vbr, __bf16* __restrict__ vbi,
    const float* __restrict__ ahyr, const float* __restrict__ ahyi,
    const __bf16* __restrict__ Wr, const float* __restrict__ br,
    const __bf16* __restrict__ Ur, const float* __restrict__ ubr,
    const __bf16* __restrict__ Wi, const float* __restrict__ bi,
    const __bf16* __restrict__ Ui, const float* __restrict__ ubi,
    const float* __restrict__ wdr, const float* __restrict__ wdi,
    const float* __restrict__ reta) {
  __shared__ __bf16 sA[4][16][NDIM];  // 32 KB: z_r, v_r, z_i, v_i row panels
  const int tid = threadIdx.x;
  const int rowBase = blockIdx.x * 16;

  {  // cooperative stage: 4 panels x 8 KB, two uint4 per thread per panel
    const __bf16* srcs[4] = {zbr, vbr, zbi, vbi};
#pragma unroll
    for (int a = 0; a < 4; ++a) {
      const uint4* g = (const uint4*)(srcs[a] + (size_t)rowBase * NDIM);
      uint4* s = (uint4*)&sA[a][0][0];
      s[tid]       = g[tid];
      s[tid + 256] = g[tid + 256];
    }
  }
  __syncthreads();

  const int lane = tid & 31;
  const int wave = tid >> 5;
  const int colBase = wave * 32;
  const int half = lane >> 4;
  const int lr = lane & 15;

  v8f acc[2][2];
  const v8f z8 = {0.f, 0.f, 0.f, 0.f, 0.f, 0.f, 0.f, 0.f};
  acc[0][0] = z8; acc[0][1] = z8; acc[1][0] = z8; acc[1][1] = z8;

  const __bf16* Bsrc[2][2] = {{Wr, Ur}, {Wi, Ui}};
#pragma unroll
  for (int g = 0; g < 2; ++g) {
#pragma unroll
    for (int t = 0; t < 2; ++t) {
      const __bf16* As = &sA[g * 2 + t][lr][half * 16];
      const __bf16* B0 = Bsrc[g][t] + (size_t)(colBase + lr) * NDIM + half * 16;
      const __bf16* B1 = B0 + (size_t)16 * NDIM;
      for (int kb = 0; kb < NDIM; kb += 32) {
        __builtin_prefetch(B0 + kb + 32, 0, 1);
        v8bf lo = *(const v8bf*)(As + kb);        // ds_load_b128
        v8bf hi = *(const v8bf*)(As + kb + 8);    // ds_load_b128
        v16bf a = __builtin_shufflevector(lo, hi, 0, 1, 2, 3, 4, 5, 6, 7,
                                          8, 9, 10, 11, 12, 13, 14, 15);
        v16bf b0 = *(const v16bf*)(B0 + kb);
        acc[g][0] = wmma_bf16(a, b0, acc[g][0]);
        v16bf b1 = *(const v16bf*)(B1 + kb);
        acc[g][1] = wmma_bf16(a, b1, acc[g][1]);
      }
    }
  }

  const float eta = softplus(reta[0]);
#pragma unroll
  for (int ct = 0; ct < 2; ++ct) {
    const int n = colBase + ct * 16 + lr;
    const float biasR = br[n] + ubr[n];
    const float biasI = bi[n] + ubi[n];
    const float dr = wdr[n] + eta;
    const float di = wdi[n];
    const float invden = 1.0f / fmaf(dr, dr, di * di);
    const v8f aR = acc[0][ct];
    const v8f aI = acc[1][ct];
#pragma unroll
    for (int j = 0; j < 8; ++j) {
      const int m = rowBase + half * 8 + j;  // C-tile: lanes 0-15 -> M=j, 16-31 -> M=8+j
      const size_t idx = (size_t)m * NDIM + n;
      const float gr = sigm(aR[j] + biasR);
      const float gi = sigm(aI[j] + biasI);
      const float Zr = zr[idx], Zi = zi[idx];
      const float Vr = vr[idx], Vi = vi[idx];
      const float nr = ahyr[idx] + eta * (Zr - Vr);
      const float ni = ahyi[idx] + eta * (Zi - Vi);
      const float xr = (nr * dr + ni * di) * invden;
      const float xi = (ni * dr - nr * di) * invden;
      const float ur_ = xr + Vr, ui_ = xi + Vi;
      const float znr = gr * ur_ - gi * ui_ + (1.f - gr) * Zr + gi * Zi;
      const float zni = gr * ui_ + gi * ur_ + (1.f - gr) * Zi - gi * Zr;
      const float vnr = Vr + xr - znr;
      const float vni = Vi + xi - zni;
      zr[idx] = znr; zi[idx] = zni;
      vr[idx] = vnr; vi[idx] = vni;
      zbr[idx] = (__bf16)znr; zbi[idx] = (__bf16)zni;
      vbr[idx] = (__bf16)vnr; vbi[idx] = (__bf16)vni;
    }
  }
}

// ---------------------------------------------------------------------------
// Final soft threshold -> interleaved complex64 output
// ---------------------------------------------------------------------------
__global__ void final_kernel(const float* __restrict__ zr, const float* __restrict__ zi,
                             const float* __restrict__ thr, const float* __restrict__ alpha,
                             float* __restrict__ out) {
  const size_t idx = (size_t)blockIdx.x * blockDim.x + threadIdx.x;
  const float R = zr[idx], I = zi[idx];
  const float amp = sqrtf(fmaf(R, R, I * I));
  const float mask = sigm(alpha[0] * (amp - thr[0]));
  out[2 * idx]     = R * mask;
  out[2 * idx + 1] = I * mask;
}

extern "C" void kernel_launch(void* const* d_in, const int* in_sizes, int n_in,
                              void* d_out, int out_size, void* d_ws, size_t ws_size,
                              hipStream_t stream) {
  const float* y_real = (const float*)d_in[0];
  const float* y_imag = (const float*)d_in[1];
  const float* A_real = (const float*)d_in[2];
  const float* A_imag = (const float*)d_in[3];
  const float* wdr    = (const float*)d_in[4];
  const float* wdi    = (const float*)d_in[5];
  const float* reta   = (const float*)d_in[6];
  const float* Wgr_w  = (const float*)d_in[7];
  const float* Wgr_b  = (const float*)d_in[8];
  const float* Ugr_w  = (const float*)d_in[9];
  const float* Ugr_b  = (const float*)d_in[10];
  const float* Wgi_w  = (const float*)d_in[11];
  const float* Wgi_b  = (const float*)d_in[12];
  const float* Ugi_w  = (const float*)d_in[13];
  const float* Ugi_b  = (const float*)d_in[14];
  const float* thr    = (const float*)d_in[15];
  const float* alpha  = (const float*)d_in[16];

  const size_t BN  = (size_t)BATCH * NDIM;
  const size_t LNN = (size_t)L_LAYERS * NDIM * NDIM;

  float* w = (float*)d_ws;
  float* ahyr  = w;
  float* ahyi  = w + 1 * BN;
  float* zr    = w + 2 * BN;
  float* zi    = w + 3 * BN;
  float* vr    = w + 4 * BN;
  float* vi    = w + 5 * BN;
  float* stats = w + 6 * BN;        // 4*N
  float* ms    = stats + 4 * NDIM;  // 4*N
  __bf16* bmem = (__bf16*)(ms + 4 * NDIM);
  __bf16* zbr = bmem;
  __bf16* zbi = bmem + 1 * BN;
  __bf16* vbr = bmem + 2 * BN;
  __bf16* vbi = bmem + 3 * BN;
  __bf16* Wbr = bmem + 4 * BN;
  __bf16* Ubr = Wbr + LNN;
  __bf16* Wbi = Ubr + LNN;
  __bf16* Ubi = Wbi + LNN;

  // one-time bf16 weight mirrors (L2-resident thereafter)
  const unsigned cvtBlocks = (unsigned)(LNN / 4 / 256);
  f32_to_bf16_kernel<<<cvtBlocks, 256, 0, stream>>>(Wgr_w, Wbr);
  f32_to_bf16_kernel<<<cvtBlocks, 256, 0, stream>>>(Ugr_w, Ubr);
  f32_to_bf16_kernel<<<cvtBlocks, 256, 0, stream>>>(Wgi_w, Wbi);
  f32_to_bf16_kernel<<<cvtBlocks, 256, 0, stream>>>(Ugi_w, Ubi);

  ahy_kernel<<<BATCH / 4, 256, 0, stream>>>(y_real, y_imag, A_real, A_imag, ahyr, ahyi);
  stats_init_kernel<<<1, NDIM, 0, stream>>>(stats);
  bn_partial_kernel<<<64, NDIM, 0, stream>>>(ahyr, ahyi, stats);
  bn_finalize_kernel<<<1, NDIM, 0, stream>>>(stats, ms);
  bn_apply_kernel<<<(unsigned)(BN / 256), 256, 0, stream>>>(ahyr, ahyi, ms, zr, zi, vr, vi,
                                                            zbr, zbi, vbr, vbi);

  for (int l = 0; l < L_LAYERS; ++l) {
    const size_t oNN = (size_t)l * NDIM * NDIM;
    const size_t oN  = (size_t)l * NDIM;
    layer_kernel<<<BATCH / 16, 256, 0, stream>>>(
        zr, zi, vr, vi, zbr, zbi, vbr, vbi, ahyr, ahyi,
        Wbr + oNN, Wgr_b + oN, Ubr + oNN, Ugr_b + oN,
        Wbi + oNN, Wgi_b + oN, Ubi + oNN, Ugi_b + oN,
        wdr + oN, wdi + oN, reta + l);
  }

  final_kernel<<<(unsigned)(BN / 256), 256, 0, stream>>>(zr, zi, thr, alpha, (float*)d_out);
}
